// conv_encoder_83528523972795
// MI455X (gfx1250) — matrix operand order repaired and checked by reading, not compiled, not run on hardware
//
#include <hip/hip_runtime.h>

// Problem constants (from the reference)
#define B_  32
#define C_  256
#define HW_ 3136          // 56*56 = 49 * 64
#define N_  1024
#define KSTAGE 64         // K elements per TDM stage (2 WMMA K-steps)
#define NSTG (HW_ / KSTAGE)  // 49 stages

typedef __attribute__((ext_vector_type(16))) __bf16        v16bf;
typedef __attribute__((ext_vector_type(8)))  float         v8f;
typedef __attribute__((ext_vector_type(4)))  unsigned int  u32x4v;
typedef __attribute__((ext_vector_type(8)))  int           i32x8v;
typedef __attribute__((ext_vector_type(4)))  int           i32x4v;

union V16 { v16bf v; u32x4v q[2]; };

__device__ __forceinline__ unsigned short f32_to_bf16_rne(float f) {
  union { float f; unsigned int u; } cv; cv.f = f;
  unsigned int u = cv.u;
  unsigned int r = u + 0x7FFFu + ((u >> 16) & 1u);   // round-to-nearest-even
  return (unsigned short)(r >> 16);
}

// Elementwise fp32 -> bf16 pre-conversion (memory bound; keeps VALU cvt work
// out of the WMMA inner loop and halves GEMM-side bytes).
__global__ void cvt_f32_to_bf16_x4(const float* __restrict__ in,
                                   unsigned short* __restrict__ out, int n4) {
  int i = blockIdx.x * blockDim.x + threadIdx.x;
  if (i >= n4) return;
  const float4 f = reinterpret_cast<const float4*>(in)[i];
  unsigned int lo = (unsigned int)f32_to_bf16_rne(f.x) |
                    ((unsigned int)f32_to_bf16_rne(f.y) << 16);
  unsigned int hi = (unsigned int)f32_to_bf16_rne(f.z) |
                    ((unsigned int)f32_to_bf16_rne(f.w) << 16);
  reinterpret_cast<uint2*>(out)[i] = make_uint2(lo, hi);
}

// Issue one 2-D TDM tile load (rows x tile_k bf16 elements) into LDS.
// Descriptor packing per CDNA5 ISA (groups 0/1; 2-D).
__device__ __forceinline__ void tdm_load_2d(unsigned lds_byte_off,
                                            const unsigned short* gptr,
                                            unsigned tile_k, unsigned tile_rows,
                                            unsigned row_stride_elems) {
  unsigned long long ga = (unsigned long long)(uintptr_t)gptr;
  u32x4v g0;
  g0[0] = 1u;                                     // count=1, user descriptor
  g0[1] = lds_byte_off;                           // D#.lds_addr
  g0[2] = (unsigned)ga;                           // global_addr[31:0]
  g0[3] = (unsigned)((ga >> 32) & 0x01FFFFFFu)    // global_addr[56:32]
          | (2u << 30);                           // type=2 ("image")
  const unsigned td0 = row_stride_elems;          // tensor_dim0 (no OOB in-tile)
  const unsigned td1 = tile_rows;                 // tensor_dim1
  i32x8v g1;
  g1[0] = (int)(1u << 16);                        // data_size=1 -> 2 bytes
  g1[1] = (int)((td0 & 0xFFFFu) << 16);           // tensor_dim0[15:0]
  g1[2] = (int)((td0 >> 16) | ((td1 & 0xFFFFu) << 16));
  g1[3] = (int)((td1 >> 16) | (tile_k << 16));    // tile_dim0 = K span
  g1[4] = (int)(tile_rows & 0xFFFFu);             // tile_dim1; tile_dim2=0
  g1[5] = (int)row_stride_elems;                  // tensor_dim0_stride[31:0]
  g1[6] = 0;                                      // stride hi, dim1_stride lo
  g1[7] = 0;
  i32x4v z4 = {};
#if __has_include(<hip/amd_detail/amd_gfx1250_TDM.h>)
  i32x8v z8 = {};
  __builtin_amdgcn_tensor_load_to_lds(g0, g1, z4, z4, z8, 0);
#else
  __builtin_amdgcn_tensor_load_to_lds(g0, g1, z4, z4, 0);
#endif
}

// Fused GEMM + weighted channel reduction.
// WG = 8 waves. Wave w owns c-chunks {w, w+8}; all waves share a 64-wide
// n-tile. A tiles: per-wave TDM. B tile: single TDM by wave 0, double-buffered.
__global__ __launch_bounds__(256)
void fused_gemm_tdm(const unsigned short* __restrict__ xb,
                    const unsigned short* __restrict__ wsb,
                    const float* __restrict__ Wd,
                    const float* __restrict__ Wb,
                    float* __restrict__ out) {
  __shared__ __align__(16) unsigned short ldsA[2][8][16 * KSTAGE]; // 32 KB
  __shared__ __align__(16) unsigned short ldsB[2][64 * KSTAGE];    // 16 KB
  __shared__ float red[8][4][32];                                  //  4 KB

  const int tid  = threadIdx.x;
  const int wave = __builtin_amdgcn_readfirstlane(tid >> 5);
  const int lane = tid & 31;
  const int col  = lane & 15;
  const int hiL  = lane >> 4;
  const int b     = blockIdx.y;
  const int nTile = blockIdx.x * 64;

  // LDS byte offsets (low 32 bits of flat LDS address) for the descriptors.
  const unsigned aOff0 = (unsigned)(uintptr_t)&ldsA[0][wave][0];
  const unsigned aOff1 = (unsigned)(uintptr_t)&ldsA[1][wave][0];
  const unsigned bOff0 = (unsigned)(uintptr_t)&ldsB[0][0];
  const unsigned bOff1 = (unsigned)(uintptr_t)&ldsB[1][0];

  const unsigned short* wsTile = wsb + (size_t)nTile * HW_;

  float part[4] = {0.f, 0.f, 0.f, 0.f};  // per-lane partial per n-chunk

  #pragma unroll 1
  for (int ci = 0; ci < 2; ++ci) {
    const int c0 = (wave + 8 * ci) * 16;
    const unsigned short* aTile = xb + (size_t)(b * C_ + c0) * HW_;

    v8f accs[4] = {v8f{}, v8f{}, v8f{}, v8f{}};

    __syncthreads();  // previous ci's LDS reads fully drained

    // Prologue: stage k=0 into buffer 0.
    tdm_load_2d(aOff0, aTile, KSTAGE, 16, HW_);
    if (wave == 0) tdm_load_2d(bOff0, wsTile, KSTAGE, 64, HW_);

    #pragma unroll 1
    for (int st = 0; st < NSTG; ++st) {
      const int p = st & 1;
      __builtin_amdgcn_s_wait_tensorcnt(0);  // my A (and B, wave 0) landed
      __syncthreads();                       // B visible to all; buf p^1 free

      if (st + 1 < NSTG) {                   // stage next K block
        const int kn = (st + 1) * KSTAGE;
        tdm_load_2d(p ? aOff0 : aOff1, aTile + kn, KSTAGE, 16, HW_);
        if (wave == 0)
          tdm_load_2d(p ? bOff0 : bOff1, wsTile + kn, KSTAGE, 64, HW_);
      }

      const unsigned short* As = &ldsA[p][wave][0];
      const unsigned short* Bs = &ldsB[p][0];

      // Preload ALL operands for this stage into distinct registers first so
      // the scheduler can stagger s_wait_dscnt instead of waiting to zero
      // before every WMMA (DS completes in order -> partial waits suffice).
      V16 a[2], bm[2][4];
      #pragma unroll
      for (int s = 0; s < 2; ++s) {
        // A 16x32 bf16: lane = row `col`, K = s*32 + {0..7,16..23} (+8 hi)
        const int ao = col * KSTAGE + s * 32 + hiL * 8;
        a[s].q[0] = *reinterpret_cast<const u32x4v*>(As + ao);
        a[s].q[1] = *reinterpret_cast<const u32x4v*>(As + ao + 16);
      }
      #pragma unroll
      for (int s = 0; s < 2; ++s) {
        #pragma unroll
        for (int j = 0; j < 4; ++j) {
          // B 32x16 bf16: lane = column `col` of n-chunk j, K = s*32 +
          // {0..7,16..23} (+8 hi).
          const int bo = (j * 16 + col) * KSTAGE + s * 32 + hiL * 8;
          bm[s][j].q[0] = *reinterpret_cast<const u32x4v*>(Bs + bo);
          bm[s][j].q[1] = *reinterpret_cast<const u32x4v*>(Bs + bo + 16);
        }
      }
      #pragma unroll
      for (int s = 0; s < 2; ++s) {
        #pragma unroll
        for (int j = 0; j < 4; ++j) {
          accs[j] = __builtin_amdgcn_wmma_f32_16x16x32_bf16(
              false, a[s].v, false, bm[s][j].v, (short)0, accs[j],
              false, false);
        }
      }
    }

    // D layout: lane = n column; VGPR v = row c0 + v + hiL*8. Weight by W_d.
    #pragma unroll
    for (int j = 0; j < 4; ++j) {
      const float* wd = Wd + (size_t)(nTile + j * 16 + col) * C_ + c0 + hiL * 8;
      const float4 w0 = reinterpret_cast<const float4*>(wd)[0];
      const float4 w1 = reinterpret_cast<const float4*>(wd)[1];
      const v8f acc = accs[j];
      part[j] += acc[0] * w0.x + acc[1] * w0.y + acc[2] * w0.z + acc[3] * w0.w
               + acc[4] * w1.x + acc[5] * w1.y + acc[6] * w1.z + acc[7] * w1.w;
    }
  }

  // Fold 8 waves x 2 lane-halves per n, add bias, store.
  red[wave][0][lane] = part[0];
  red[wave][1][lane] = part[1];
  red[wave][2][lane] = part[2];
  red[wave][3][lane] = part[3];
  __syncthreads();

  if (tid < 64) {
    const int j  = tid >> 4;
    const int cc = tid & 15;
    float s = 0.f;
    #pragma unroll
    for (int w = 0; w < 8; ++w) s += red[w][j][cc] + red[w][j][cc + 16];
    const int n = nTile + j * 16 + cc;
    out[(size_t)b * N_ + n] = s + Wb[n];
  }
}

extern "C" void kernel_launch(void* const* d_in, const int* in_sizes, int n_in,
                              void* d_out, int out_size, void* d_ws, size_t ws_size,
                              hipStream_t stream) {
  const float* x  = (const float*)d_in[0];   // (B, C, H, W)
  const float* Ws = (const float*)d_in[1];   // (N, H, W)
  const float* Wd = (const float*)d_in[2];   // (N, C, 1, 1) -> (N, C)
  const float* Wb = (const float*)d_in[3];   // (1, N)
  float* out = (float*)d_out;                // (B, N)

  // Workspace: bf16 copies of x and W_s (51.4 MB + 6.4 MB = 57.8 MB).
  unsigned short* xb  = (unsigned short*)d_ws;
  unsigned short* wsb = xb + (size_t)B_ * C_ * HW_;

  {
    const int n4 = (B_ * C_ * HW_) / 4;      // divisible by 256
    cvt_f32_to_bf16_x4<<<n4 / 256, 256, 0, stream>>>(x, xb, n4);
  }
  {
    const int n4 = (N_ * HW_) / 4;           // divisible by 256
    cvt_f32_to_bf16_x4<<<n4 / 256, 256, 0, stream>>>(Ws, wsb, n4);
  }

  dim3 grid(N_ / 64, B_);                    // (16, 32) workgroups
  fused_gemm_tdm<<<grid, 256, 0, stream>>>(xb, wsb, Wd, Wb, out);
}
